// LSTMFC_32856499814963
// MI455X (gfx1250) — matrix operand order repaired and checked by reading, not compiled, not run on hardware
//
#include <hip/hip_runtime.h>
#include <cstdint>
#include <cstddef>

// ---------------------------------------------------------------------------
// Problem dims (compile-time)
// ---------------------------------------------------------------------------
#define B_  64
#define T_  1024
#define I_  128
#define H_  256
#define G4_ 1024   // 4*H
#define O_  128

typedef _Float16 h8   __attribute__((ext_vector_type(8)));
typedef _Float16 v16h __attribute__((ext_vector_type(16)));
typedef float    v8f  __attribute__((ext_vector_type(8)));

static __device__ __forceinline__ v16h cat16(h8 lo, h8 hi) {
  return __builtin_shufflevector(lo, hi, 0, 1, 2, 3, 4, 5, 6, 7,
                                 8, 9, 10, 11, 12, 13, 14, 15);
}

static __device__ __forceinline__ float sigf(float x) {
  return 1.0f / (1.0f + __expf(-x));
}

static __device__ __forceinline__ v8f wmma_f16(v16h a, v16h b, v8f c) {
  return __builtin_amdgcn_wmma_f32_16x16x32_f16(
      /*neg_a=*/false, a, /*neg_b=*/false, b,
      /*c_mod=*/(short)0, c, /*reuse_a=*/false, /*reuse_b=*/false);
}

// ---------------------------------------------------------------------------
// Small prep kernels
// ---------------------------------------------------------------------------
__global__ void k_f32_to_f16(const float* __restrict__ src,
                             _Float16* __restrict__ dst, int n) {
  int i = blockIdx.x * blockDim.x + threadIdx.x;
  if (i < n) dst[i] = (_Float16)src[i];
}

__global__ void k_bias_sum(const float* __restrict__ a,
                           const float* __restrict__ b,
                           float* __restrict__ o, int n) {
  int i = blockIdx.x * blockDim.x + threadIdx.x;
  if (i < n) o[i] = a[i] + b[i];
}

// ---------------------------------------------------------------------------
// Phase 1: Gx[t] = x_t @ W_ih^T + (b_ih + b_hh), for all t in parallel.
// Output stored in WMMA C-fragment order (f16):
//   Gx[((t*256 + tile)*32 + lane)*8 + i],  tile = mtile*64 + ntile
// so the recurrence kernel re-loads C fragments with one b128 per lane.
// One wave per 16x16 output tile; K = 128 = 4 x wmma_f32_16x16x32_f16.
// Gx is 128 MB f16 -> resident in the 192 MB global L2.
// ---------------------------------------------------------------------------
__global__ __launch_bounds__(256)
void k_xgemm(const float* __restrict__ x,        // [B, T, I]
             const _Float16* __restrict__ Wih,   // [4H, I] f16
             const float* __restrict__ bias,     // [4H] = b_ih + b_hh
             _Float16* __restrict__ Gx) {        // fragment layout, f16
  const int lane = threadIdx.x & 31;
  const int gw = blockIdx.x * (blockDim.x >> 5) + (threadIdx.x >> 5);
  const int t = gw >> 8;            // 1024 t values
  if (t >= T_) return;
  const int rem = gw & 255;
  const int mtile = rem >> 6;       // 0..3  (batch tiles of 16)
  const int ntile = rem & 63;       // 0..63 (gate-column tiles of 16)
  const int col = lane & 15;
  const int hi = lane >> 4;

  const int n = ntile * 16 + col;                  // gate column (B operand)
  const int brow = mtile * 16 + col;               // batch row (A operand)

  v8f c;
  const float bv = bias[n];
#pragma unroll
  for (int i = 0; i < 8; ++i) c[i] = bv;

  const float* __restrict__ xrow = x + ((size_t)brow * T_ + t) * I_;
  const _Float16* __restrict__ wrow = Wih + (size_t)n * I_;

#pragma unroll
  for (int kk = 0; kk < 4; ++kk) {
    // A: row = brow; lanes<16 hold K in {0..7,16..23}, lanes>=16 {8..15,24..31}
    const int ka = kk * 32 + hi * 8;
    h8 alo, ahi;
#pragma unroll
    for (int j = 0; j < 8; ++j) {
      alo[j] = (_Float16)xrow[ka + j];
      ahi[j] = (_Float16)xrow[ka + 16 + j];
    }
    v16h a = cat16(alo, ahi);
    // B: column n of W_ih^T == row n of W_ih (two 8-f16 runs per lane, same K split as A)
    const int kb = kk * 32 + hi * 8;
    v16h b = cat16(*(const h8*)(wrow + kb), *(const h8*)(wrow + kb + 16));
    c = wmma_f16(a, b, c);
  }

  h8 r;
#pragma unroll
  for (int i = 0; i < 8; ++i) r[i] = (_Float16)c[i];
  const size_t off =
      (((size_t)t * 256 + (size_t)(mtile * 64 + ntile)) * 32 + lane) * 8;
  *(h8*)(Gx + off) = r;
}

// ---------------------------------------------------------------------------
// Phase 2: persistent single-workgroup recurrence (1 WGP, 32 wave32 waves).
// Only the gate GEMM + cell update are on the serial path; h(t) is spilled to
// global (f16, L2-resident) and the output projection runs afterwards in
// parallel over all t.
//   h ping-pong (2 x 64x256 f16 = 64 KB) in LDS -> ONE barrier per step
//   c state lives in VGPRs per wave for all 1024 steps
//   W_hh (f16, 512 KB) streams from L2 each step
//   Gx fragments are loaded at the top of the step but consumed only AFTER
//   the K-loop (acc starts at 0, Gx added at the end), so their L2 latency
//   is hidden behind the 64-WMMA gate GEMM instead of stalling its start.
//   gate order (torch LSTMCell): i, f, g, o  -> columns q*256 + hidx
// Unit map: wave w -> h-tile jh = w>>1, batch tiles {2*(w&1), 2*(w&1)+1}
// ---------------------------------------------------------------------------
__global__ __launch_bounds__(1024)
void k_lstm(const _Float16* __restrict__ Gx,    // fragment layout
            const _Float16* __restrict__ Whh,   // [4H, H] f16
            _Float16* __restrict__ Hseq) {      // [T, B, H] f16 out
  __shared__ __align__(16) _Float16 hbuf[2][B_ * H_];  // 2 x 32 KB

  const int tid = threadIdx.x;
  const int lane = tid & 31;
  const int w = tid >> 5;
  const int col = lane & 15;
  const int hi = lane >> 4;

  for (int i = tid; i < B_ * H_; i += 1024) hbuf[0][i] = (_Float16)0.0f;

  const int jh = w >> 1;            // 0..15  h-tile
  const int mt0 = (w & 1) * 2;      // batch tiles {mt0, mt0+1}

  float creg[2][8];
#pragma unroll
  for (int u = 0; u < 2; ++u)
#pragma unroll
    for (int i = 0; i < 8; ++i) creg[u][i] = 0.0f;

  __syncthreads();

  for (int t = 0; t < T_; ++t) {
    const _Float16* __restrict__ hrd = hbuf[t & 1];
    _Float16* __restrict__ hwr = hbuf[(t & 1) ^ 1];

    // ---- issue Gx fragment loads now, consume after the K-loop ----------
    h8 gfrag[2][4];
#pragma unroll
    for (int u = 0; u < 2; ++u) {
      const int mt = mt0 + u;
#pragma unroll
      for (int q = 0; q < 4; ++q) {
        const int tile = mt * 64 + q * 16 + jh;
        gfrag[u][q] =
            *(const h8*)(Gx + (((size_t)t * 256 + tile) * 32 + lane) * 8);
      }
    }

    // ---- h @ W_hh^T ------------------------------------------------------
    v8f acc[2][4];
#pragma unroll
    for (int u = 0; u < 2; ++u)
#pragma unroll
      for (int q = 0; q < 4; ++q)
#pragma unroll
        for (int i = 0; i < 8; ++i) acc[u][q][i] = 0.0f;

#pragma unroll 4
    for (int kk = 0; kk < 8; ++kk) {       // K = 256 = 8 x 32
      v16h a[2];
#pragma unroll
      for (int u = 0; u < 2; ++u) {
        const _Float16* hp =
            hrd + ((mt0 + u) * 16 + col) * H_ + kk * 32 + hi * 8;
        a[u] = cat16(*(const h8*)hp, *(const h8*)(hp + 16));
      }
#pragma unroll
      for (int q = 0; q < 4; ++q) {
        const int n = q * 256 + jh * 16 + col;
        const _Float16* wr = Whh + (size_t)n * H_ + kk * 32 + hi * 8;
        v16h b = cat16(*(const h8*)wr, *(const h8*)(wr + 16));
        acc[0][q] = wmma_f16(a[0], b, acc[0][q]);
        acc[1][q] = wmma_f16(a[1], b, acc[1][q]);
      }
    }

    // ---- add the precomputed input-gate fragment (latency now covered) --
#pragma unroll
    for (int u = 0; u < 2; ++u)
#pragma unroll
      for (int q = 0; q < 4; ++q)
#pragma unroll
        for (int i = 0; i < 8; ++i) acc[u][q][i] += (float)gfrag[u][q][i];

    // ---- LSTM cell elementwise: update c (regs), h -> LDS(pong) + global
#pragma unroll
    for (int u = 0; u < 2; ++u) {
      const int mt = mt0 + u;
#pragma unroll
      for (int i = 0; i < 8; ++i) {
        const float ig = sigf(acc[u][0][i]);
        const float fg = sigf(acc[u][1][i]);
        const float gg = tanhf(acc[u][2][i]);
        const float og = sigf(acc[u][3][i]);
        const float cv = fg * creg[u][i] + ig * gg;
        creg[u][i] = cv;
        const _Float16 hv = (_Float16)(og * tanhf(cv));
        const int brow = mt * 16 + i + 8 * hi;
        const int hidx = jh * 16 + col;
        hwr[brow * H_ + hidx] = hv;
        Hseq[((size_t)t * B_ + brow) * H_ + hidx] = hv;
      }
    }

    __syncthreads();  // h(t) visible; also protects reuse of hrd next step
  }
}

// ---------------------------------------------------------------------------
// Phase 3: y[t] = h(t) @ W_out^T + b_out for all t in parallel.
// One wave per 16x16 tile; tiles = T * (B/16) * (O/16) = 32768.
// ---------------------------------------------------------------------------
__global__ __launch_bounds__(256)
void k_ygemm(const _Float16* __restrict__ Hseq,  // [T, B, H] f16
             const _Float16* __restrict__ Wout,  // [O, H] f16
             const float* __restrict__ bout,     // [O]
             float* __restrict__ out) {          // [B, T, O] f32
  const int lane = threadIdx.x & 31;
  const int gw = blockIdx.x * (blockDim.x >> 5) + (threadIdx.x >> 5);
  const int t = gw >> 5;
  if (t >= T_) return;
  const int rem = gw & 31;
  const int ymt = rem >> 3;         // batch tile 0..3
  const int yot = rem & 7;          // out tile   0..7
  const int col = lane & 15;
  const int hi = lane >> 4;

  const int o = yot * 16 + col;

  v8f c;
  const float bv = bout[o];
#pragma unroll
  for (int i = 0; i < 8; ++i) c[i] = bv;

  const _Float16* __restrict__ hrow =
      Hseq + ((size_t)t * B_ + ymt * 16 + col) * H_;
  const _Float16* __restrict__ wrow = Wout + (size_t)o * H_;

#pragma unroll 2
  for (int kk = 0; kk < 8; ++kk) {       // K = 256
    const int ka = kk * 32 + hi * 8;
    v16h a = cat16(*(const h8*)(hrow + ka), *(const h8*)(hrow + ka + 16));
    const int kb = kk * 32 + hi * 8;
    v16h b = cat16(*(const h8*)(wrow + kb), *(const h8*)(wrow + kb + 16));
    c = wmma_f16(a, b, c);
  }

#pragma unroll
  for (int i = 0; i < 8; ++i) {
    const int brow = ymt * 16 + i + 8 * hi;
    out[((size_t)brow * T_ + t) * O_ + o] = c[i];
  }
}

// ---------------------------------------------------------------------------
// Launcher. Input order: input, W_ih, W_hh, b_ih, b_hh, W_out, b_out, silence
// ---------------------------------------------------------------------------
extern "C" void kernel_launch(void* const* d_in, const int* in_sizes, int n_in,
                              void* d_out, int out_size, void* d_ws,
                              size_t ws_size, hipStream_t stream) {
  (void)in_sizes; (void)n_in; (void)out_size; (void)ws_size;

  const float* x     = (const float*)d_in[0];
  const float* W_ih  = (const float*)d_in[1];
  const float* W_hh  = (const float*)d_in[2];
  const float* b_ih  = (const float*)d_in[3];
  const float* b_hh  = (const float*)d_in[4];
  const float* W_out = (const float*)d_in[5];
  const float* b_out = (const float*)d_in[6];
  float* out = (float*)d_out;

  // Workspace layout (256B aligned blocks)
  uint8_t* ws = (uint8_t*)d_ws;
  size_t off = 0;
  auto take = [&](size_t bytes) {
    uint8_t* p = ws + off;
    off += (bytes + 255) & ~(size_t)255;
    return p;
  };
  _Float16* Wih16  = (_Float16*)take((size_t)G4_ * I_ * 2);      // 256 KB
  _Float16* Whh16  = (_Float16*)take((size_t)G4_ * H_ * 2);      // 512 KB
  _Float16* Wout16 = (_Float16*)take((size_t)O_ * H_ * 2);       //  64 KB
  float*    bias   = (float*)take((size_t)G4_ * 4);              //   4 KB
  _Float16* Gx     = (_Float16*)take((size_t)T_ * G4_ * B_ * 2); // 128 MB
  _Float16* Hseq   = (_Float16*)take((size_t)T_ * B_ * H_ * 2);  //  32 MB

  // Weight conversion + fused bias
  k_f32_to_f16<<<(G4_ * I_ + 255) / 256, 256, 0, stream>>>(W_ih, Wih16,
                                                           G4_ * I_);
  k_f32_to_f16<<<(G4_ * H_ + 255) / 256, 256, 0, stream>>>(W_hh, Whh16,
                                                           G4_ * H_);
  k_f32_to_f16<<<(O_ * H_ + 255) / 256, 256, 0, stream>>>(W_out, Wout16,
                                                          O_ * H_);
  k_bias_sum<<<(G4_ + 255) / 256, 256, 0, stream>>>(b_ih, b_hh, bias, G4_);

  // Parallel input-gate GEMM: T*4*64 = 262144 tiles, 1 wave each, 8 waves/block
  k_xgemm<<<262144 / 8, 256, 0, stream>>>(x, Wih16, bias, Gx);

  // Persistent recurrence: one 1024-thread workgroup (32 waves on one WGP)
  k_lstm<<<1, 1024, 0, stream>>>(Gx, Whh16, Hseq);

  // Parallel output projection: 32768 tiles, 1 wave each, 8 waves/block
  k_ygemm<<<32768 / 8, 256, 0, stream>>>(Hseq, Wout16, b_out, out);
}
